// MultiHeadSelfAttention_5497558139469
// MI455X (gfx1250) — matrix operand order repaired and checked by reading, not compiled, not run on hardware
//
#include <hip/hip_runtime.h>
#include <cstdint>

typedef __attribute__((ext_vector_type(16))) __bf16 v16bf;
typedef __attribute__((ext_vector_type(8)))  float  v8f;

union FragAB {
    v16bf    bf;
    uint32_t u[8];
};

__device__ __forceinline__ uint16_t f2bf(float x) {
    union { float f; uint32_t u; } c;
    c.f = x;
    uint32_t r = (c.u + 0x7FFFu + ((c.u >> 16) & 1u)) >> 16;  // round-to-nearest-even
    return (uint16_t)r;
}

static constexpr int BB = 4;
static constexpr int SS = 2048;
static constexpr int EE = 1024;
static constexpr int M_TOK = BB * SS;  // 8192

// ---------------------------------------------------------------------------
// Elementwise casts
// ---------------------------------------------------------------------------
__global__ void cast_f32_to_bf16(const float* __restrict__ src,
                                 uint16_t* __restrict__ dst, int n) {
    int i = blockIdx.x * blockDim.x + threadIdx.x;
    if (i < n) dst[i] = f2bf(src[i]);
}

// W is [N,K] row-major fp32 (nn.Linear weight); produce Wt[k][n] = W[n][k] bf16
__global__ void transpose_cast_f32_to_bf16(const float* __restrict__ W,
                                           uint16_t* __restrict__ Wt,
                                           int N, int K) {
    int i = blockIdx.x * blockDim.x + threadIdx.x;
    if (i < N * K) {
        int n = i / K;
        int k = i - n * K;
        Wt[(size_t)k * N + n] = f2bf(W[i]);
    }
}

// ---------------------------------------------------------------------------
// Generic WMMA GEMM: Y[M,N] = A[M,K](bf16) @ Wt[K,N](bf16) + bias[N]
// OUT_MODE 0: bf16 row-major    1: bf16 transposed per batch ([B][N][S])
//          2: fp32 row-major
// Block = 256 threads (8 waves). The 16-row A strip (16 x K bf16 = 32 KB,
// contiguous in global memory) is staged once into LDS with CDNA5 async
// loads (global_load_async_to_lds_b128, ASYNCcnt). Each wave then computes
// a 16x64 output tile (4 WMMA accumulators), reading A fragments from LDS.
// Requires K == EE (all call sites satisfy this).
// ---------------------------------------------------------------------------
template <int OUT_MODE>
__global__ void __launch_bounds__(256)
gemm_wmma_bf16(const uint16_t* __restrict__ A,
               const uint16_t* __restrict__ Wt,
               const float* __restrict__ bias,
               void* __restrict__ out,
               int Mdim, int Ndim, int Kdim, int Sdim) {
    __shared__ __align__(16) uint16_t shA[16 * EE];  // 32 KB A strip

    const int lane = threadIdx.x & 31;
    const int wave = threadIdx.x >> 5;
    const int half = lane >> 4;
    const int r15  = lane & 15;

    const int n0 = blockIdx.x * 512 + wave * 64;
    const int m0 = blockIdx.y * 16;

    // ---- async-stage the contiguous 16 x K A strip into LDS --------------
    {
        const char*    astrip  = (const char*)(A + (size_t)m0 * Kdim);
        const unsigned ldsBase = (unsigned)(size_t)(&shA[0]);  // low 32 bits of
                                                               // flat addr == LDS offset
#pragma unroll
        for (int it = 0; it < 8; ++it) {
            const int      idx = (threadIdx.x + 256 * it) * 16;  // byte offset
            const unsigned lo  = ldsBase + idx;
            const uint64_t ga  = (uint64_t)(size_t)astrip + idx;
            asm volatile("global_load_async_to_lds_b128 %0, %1, off"
                         :: "v"(lo), "v"(ga) : "memory");
        }
        asm volatile("s_wait_asynccnt 0x0" ::: "memory");
    }
    __syncthreads();

    v8f c[4];
    v8f zero = {};
#pragma unroll
    for (int nt = 0; nt < 4; ++nt) c[nt] = zero;

    const int kB = r15 + 16 * half;  // B-frag K row per lane

    for (int k0 = 0; k0 < Kdim; k0 += 32) {
        FragAB a;
#pragma unroll
        for (int j = 0; j < 8; ++j) {
            int koff = (j < 4 ? 2 * j : 16 + 2 * (j - 4)) + 8 * half;
            a.u[j] = *reinterpret_cast<const uint32_t*>(&shA[r15 * EE + k0 + koff]);
        }
        const uint16_t* wrow = &Wt[(size_t)k0 * Ndim + n0 + r15];
#pragma unroll
        for (int nt = 0; nt < 4; ++nt) {
            FragAB b;
#pragma unroll
            for (int v = 0; v < 8; ++v) {
                int kk = (v < 4 ? 2 * v : 16 + 2 * (v - 4)) + 8 * half;
                b.u[v] = (uint32_t)wrow[(size_t)kk * Ndim + nt * 16] | ((uint32_t)wrow[(size_t)(kk + 1) * Ndim + nt * 16] << 16);
            }
            c[nt] = __builtin_amdgcn_wmma_f32_16x16x32_bf16(false, a.bf, false, b.bf,
                                                            (short)0, c[nt], false, false);
        }
    }

#pragma unroll
    for (int nt = 0; nt < 4; ++nt) {
        const int   n  = n0 + nt * 16 + r15;
        const float bn = bias[n];
#pragma unroll
        for (int r = 0; r < 8; ++r) {
            int   m = m0 + r + 8 * half;
            float y = c[nt][r] + bn;
            if constexpr (OUT_MODE == 0) {
                ((uint16_t*)out)[(size_t)m * Ndim + n] = f2bf(y);
            } else if constexpr (OUT_MODE == 1) {
                int bidx = m / Sdim;
                int sidx = m - bidx * Sdim;
                ((uint16_t*)out)[((size_t)bidx * Ndim + n) * Sdim + sidx] = f2bf(y);
            } else {
                ((float*)out)[(size_t)m * Ndim + n] = y;
            }
        }
    }
}

// ---------------------------------------------------------------------------
// Fused attention: one block per (16-query strip, batch).
// Stage 1: scores = Q Kt (16x2048, per wave a 16x256 strip in registers),
//          softmax across the block (shuffle + LDS reductions),
//          normalized P written to LDS as bf16 [16][2048] (64 KB).
// Stage 2: O = P V, wave owns 128 output columns; A-frags from LDS.
// ---------------------------------------------------------------------------
__global__ void __launch_bounds__(256)
attention_fused(const uint16_t* __restrict__ Qb,   // [B*S, E] bf16
                const uint16_t* __restrict__ Ktb,  // [B][E][S] bf16
                const uint16_t* __restrict__ Vb,   // [B*S, E] bf16
                uint16_t* __restrict__ Ab,         // [B*S, E] bf16
                float scale) {
    __shared__ __align__(16) uint16_t shP[16 * SS];          // 64 KB
    float* redF = reinterpret_cast<float*>(shP);             // aliased scratch

    const int lane = threadIdx.x & 31;
    const int wave = threadIdx.x >> 5;
    const int half = lane >> 4;
    const int r15  = lane & 15;

    const int b  = blockIdx.y;
    const int m0 = blockIdx.x * 16;

    const size_t qrow   = (size_t)(b * SS + m0 + r15) * EE;
    const size_t ktBase = (size_t)b * EE * SS;
    const int    kB     = r15 + 16 * half;

    // ---- Stage 1: scores --------------------------------------------------
    v8f zero = {};
    v8f sc[16];
#pragma unroll
    for (int t = 0; t < 16; ++t) sc[t] = zero;

    for (int k0 = 0; k0 < EE; k0 += 32) {
        FragAB a;
#pragma unroll
        for (int j = 0; j < 8; ++j) {
            int koff = (j < 4 ? 2 * j : 16 + 2 * (j - 4)) + 8 * half;
            a.u[j] = *reinterpret_cast<const uint32_t*>(&Qb[qrow + k0 + koff]);
        }
        const uint16_t* krow = &Ktb[ktBase + (size_t)k0 * SS + wave * 256 + r15];
#pragma unroll
        for (int t = 0; t < 16; ++t) {
            FragAB bm;
#pragma unroll
            for (int v = 0; v < 8; ++v) {
                int kk = (v < 4 ? 2 * v : 16 + 2 * (v - 4)) + 8 * half;
                bm.u[v] = (uint32_t)krow[(size_t)kk * SS + t * 16] | ((uint32_t)krow[(size_t)(kk + 1) * SS + t * 16] << 16);
            }
            sc[t] = __builtin_amdgcn_wmma_f32_16x16x32_bf16(false, a.bf, false, bm.bf,
                                                            (short)0, sc[t], false, false);
        }
    }

    // ---- softmax: row max -------------------------------------------------
    float lmax[8];
#pragma unroll
    for (int i = 0; i < 8; ++i) lmax[i] = -3.0e38f;
#pragma unroll
    for (int t = 0; t < 16; ++t)
#pragma unroll
        for (int i = 0; i < 8; ++i) {
            sc[t][i] *= scale;
            lmax[i] = fmaxf(lmax[i], sc[t][i]);
        }
#pragma unroll
    for (int mask = 1; mask <= 8; mask <<= 1)
#pragma unroll
        for (int i = 0; i < 8; ++i)
            lmax[i] = fmaxf(lmax[i], __shfl_xor(lmax[i], mask, 32));

    if (r15 == 0) {
#pragma unroll
        for (int i = 0; i < 8; ++i) redF[wave * 16 + i + 8 * half] = lmax[i];
    }
    __syncthreads();
    float rmax[8];
#pragma unroll
    for (int i = 0; i < 8; ++i) {
        float m = redF[i + 8 * half];
#pragma unroll
        for (int w2 = 1; w2 < 8; ++w2)
            m = fmaxf(m, redF[w2 * 16 + i + 8 * half]);
        rmax[i] = m;
    }
    __syncthreads();

    // ---- softmax: exp + row sum ------------------------------------------
    float lsum[8];
#pragma unroll
    for (int i = 0; i < 8; ++i) lsum[i] = 0.0f;
#pragma unroll
    for (int t = 0; t < 16; ++t)
#pragma unroll
        for (int i = 0; i < 8; ++i) {
            float p = __expf(sc[t][i] - rmax[i]);
            sc[t][i] = p;
            lsum[i] += p;
        }
#pragma unroll
    for (int mask = 1; mask <= 8; mask <<= 1)
#pragma unroll
        for (int i = 0; i < 8; ++i)
            lsum[i] += __shfl_xor(lsum[i], mask, 32);

    if (r15 == 0) {
#pragma unroll
        for (int i = 0; i < 8; ++i) redF[wave * 16 + i + 8 * half] = lsum[i];
    }
    __syncthreads();
    float rinv[8];
#pragma unroll
    for (int i = 0; i < 8; ++i) {
        float s = 0.0f;
#pragma unroll
        for (int w2 = 0; w2 < 8; ++w2)
            s += redF[w2 * 16 + i + 8 * half];
        rinv[i] = 1.0f / s;
    }
    __syncthreads();  // reduction scratch dead; safe to overwrite with P

    // ---- write normalized P to LDS ---------------------------------------
#pragma unroll
    for (int t = 0; t < 16; ++t) {
        int n = wave * 256 + t * 16 + r15;
#pragma unroll
        for (int i = 0; i < 8; ++i) {
            int m = i + 8 * half;
            shP[m * SS + n] = f2bf(sc[t][i] * rinv[i]);
        }
    }
    __syncthreads();

    // ---- Stage 2: O = P @ V ----------------------------------------------
    v8f oacc[8];
#pragma unroll
    for (int nt = 0; nt < 8; ++nt) oacc[nt] = zero;

    for (int k0 = 0; k0 < SS; k0 += 32) {
        FragAB a;
#pragma unroll
        for (int j = 0; j < 8; ++j) {
            int koff = (j < 4 ? 2 * j : 16 + 2 * (j - 4)) + 8 * half;
            a.u[j] = *reinterpret_cast<const uint32_t*>(&shP[r15 * SS + k0 + koff]);
        }
        const size_t vrow = (size_t)(b * SS + k0) * EE + wave * 128 + r15;
#pragma unroll
        for (int nt = 0; nt < 8; ++nt) {
            FragAB bm;
#pragma unroll
            for (int v = 0; v < 8; ++v) {
                int kk = (v < 4 ? 2 * v : 16 + 2 * (v - 4)) + 8 * half;
                bm.u[v] = (uint32_t)Vb[vrow + (size_t)kk * EE + nt * 16] | ((uint32_t)Vb[vrow + (size_t)(kk + 1) * EE + nt * 16] << 16);
            }
            oacc[nt] = __builtin_amdgcn_wmma_f32_16x16x32_bf16(false, a.bf, false, bm.bf,
                                                               (short)0, oacc[nt], false, false);
        }
    }

#pragma unroll
    for (int nt = 0; nt < 8; ++nt) {
        int n = wave * 128 + nt * 16 + r15;
#pragma unroll
        for (int i = 0; i < 8; ++i) {
            int m = m0 + i + 8 * half;
            Ab[(size_t)(b * SS + m) * EE + n] = f2bf(oacc[nt][i]);
        }
    }
}

// ---------------------------------------------------------------------------
extern "C" void kernel_launch(void* const* d_in, const int* in_sizes, int n_in,
                              void* d_out, int out_size, void* d_ws, size_t ws_size,
                              hipStream_t stream) {
    (void)in_sizes; (void)n_in; (void)out_size; (void)ws_size;

    const float* x  = (const float*)d_in[0];
    const float* Wq = (const float*)d_in[1];
    const float* bq = (const float*)d_in[2];
    const float* Wk = (const float*)d_in[3];
    const float* bk = (const float*)d_in[4];
    const float* Wv = (const float*)d_in[5];
    const float* bv = (const float*)d_in[6];
    const float* Wo = (const float*)d_in[7];
    const float* bo = (const float*)d_in[8];
    float* out = (float*)d_out;

    char* ws = (char*)d_ws;
    size_t off = 0;
    uint16_t* xb  = (uint16_t*)(ws + off); off += (size_t)M_TOK * EE * 2;  // 16 MB
    uint16_t* Wqt = (uint16_t*)(ws + off); off += (size_t)EE * EE * 2;     //  2 MB
    uint16_t* Wkt = (uint16_t*)(ws + off); off += (size_t)EE * EE * 2;
    uint16_t* Wvt = (uint16_t*)(ws + off); off += (size_t)EE * EE * 2;
    uint16_t* Wot = (uint16_t*)(ws + off); off += (size_t)EE * EE * 2;
    uint16_t* Qb  = (uint16_t*)(ws + off); off += (size_t)M_TOK * EE * 2;  // 16 MB
    uint16_t* Ktb = (uint16_t*)(ws + off); off += (size_t)M_TOK * EE * 2;  // 16 MB (as [B][E][S])
    uint16_t* Vb  = (uint16_t*)(ws + off); off += (size_t)M_TOK * EE * 2;  // 16 MB
    uint16_t* Ab  = xb;  // x is dead after the QKV projections; reuse its region

    const int threads = 256;

    // Precision staging: fp32 -> bf16 (weights transposed for B-fragment loads)
    cast_f32_to_bf16<<<(M_TOK * EE) / threads, threads, 0, stream>>>(x, xb, M_TOK * EE);
    transpose_cast_f32_to_bf16<<<(EE * EE) / threads, threads, 0, stream>>>(Wq, Wqt, EE, EE);
    transpose_cast_f32_to_bf16<<<(EE * EE) / threads, threads, 0, stream>>>(Wk, Wkt, EE, EE);
    transpose_cast_f32_to_bf16<<<(EE * EE) / threads, threads, 0, stream>>>(Wv, Wvt, EE, EE);
    transpose_cast_f32_to_bf16<<<(EE * EE) / threads, threads, 0, stream>>>(Wo, Wot, EE, EE);

    // QKV projections (K written transposed per batch: [B][E][S])
    dim3 gproj(EE / 512, M_TOK / 16);
    gemm_wmma_bf16<0><<<gproj, threads, 0, stream>>>(xb, Wqt, bq, Qb, M_TOK, EE, EE, SS);
    gemm_wmma_bf16<1><<<gproj, threads, 0, stream>>>(xb, Wkt, bk, Ktb, M_TOK, EE, EE, SS);
    gemm_wmma_bf16<0><<<gproj, threads, 0, stream>>>(xb, Wvt, bv, Vb, M_TOK, EE, EE, SS);

    // Fused scores -> softmax -> P@V
    dim3 gattn(SS / 16, BB);
    const float scale = 0.03125f;  // 1/sqrt(1024)
    attention_fused<<<gattn, threads, 0, stream>>>(Qb, Ktb, Vb, Ab, scale);

    // Output projection, fp32 result
    gemm_wmma_bf16<2><<<gproj, threads, 0, stream>>>(Ab, Wot, bo, out, M_TOK, EE, EE, SS);
}
